// EncoderLayer_88227218194924
// MI455X (gfx1250) — compile-verified, matrix operand order checked by reading, not run on hardware
//
#include <hip/hip_runtime.h>
#include <hip/hip_bf16.h>

// ---------------------------------------------------------------------------
// Types for CDNA5 WMMA (wave32)
// ---------------------------------------------------------------------------
typedef __attribute__((ext_vector_type(8)))  __bf16 v8bf;
typedef __attribute__((ext_vector_type(16))) __bf16 v16bf;
typedef __attribute__((ext_vector_type(8)))  float  v8f;

// Problem constants (fixed by the reference)
static constexpr int Bc  = 2;
static constexpr int Sc  = 2048;
static constexpr int Dc  = 1024;
static constexpr int Hc  = 16;
static constexpr int DKc = 64;
static constexpr int Mrows = Bc * Sc;      // 4096
static constexpr int D4    = 4 * Dc;       // 4096

// ---------------------------------------------------------------------------
// bf16 helpers (bit-level, RNE)
// ---------------------------------------------------------------------------
__device__ __forceinline__ unsigned short f32_to_bf16(float f) {
    unsigned int u = __float_as_uint(f);
    u += 0x7FFFu + ((u >> 16) & 1u);
    return (unsigned short)(u >> 16);
}
__device__ __forceinline__ float bf16_to_f32(unsigned short h) {
    return __uint_as_float(((unsigned int)h) << 16);
}

// ---------------------------------------------------------------------------
// Fragment loader. Produces the 16-bit 16x32 A-layout (and, symmetrically, the
// 32x16 B-layout when fed a transposed matrix): lane L(<16) holds row base+L
// with K = k0+0..7 and k0+16..23; lane L+16 holds the same row with
// K = k0+8..15 and k0+24..31.  Two contiguous 16-byte loads per lane.
// ---------------------------------------------------------------------------
__device__ __forceinline__ v16bf load_frag(const unsigned short* p, int ld,
                                           int base, int k0) {
    int lane = threadIdx.x & 31;
    const unsigned short* q =
        p + (size_t)(base + (lane & 15)) * (size_t)ld + k0 + ((lane & 16) ? 8 : 0);
    v8bf lo = *reinterpret_cast<const v8bf*>(q);
    v8bf hi = *reinterpret_cast<const v8bf*>(q + 16);
    return __builtin_shufflevector(lo, hi, 0, 1, 2, 3, 4, 5, 6, 7,
                                   8, 9, 10, 11, 12, 13, 14, 15);
}

__device__ __forceinline__ v8f wmma_bf16(v16bf a, v16bf b, v8f c) {
    return __builtin_amdgcn_wmma_f32_16x16x32_bf16(false, a, false, b,
                                                   (short)0, c, false, false);
}

// ---------------------------------------------------------------------------
// Weight convert + transpose: out[n*K + k] = bf16(in[k*N + n])
// ---------------------------------------------------------------------------
__global__ __launch_bounds__(256) void convT_kernel(const float* __restrict__ in,
                                                    unsigned short* __restrict__ out,
                                                    int K, int N) {
    size_t i = (size_t)blockIdx.x * 256 + threadIdx.x;
    if (i < (size_t)K * N) {
        int n = (int)(i / K);
        int k = (int)(i % K);
        out[i] = f32_to_bf16(in[(size_t)k * N + n]);
    }
}

// ---------------------------------------------------------------------------
// LayerNorm (f32 in, bf16 out), one block per row of D=1024
// ---------------------------------------------------------------------------
__global__ __launch_bounds__(256) void layernorm_bf16_kernel(
    const float* __restrict__ x, const float* __restrict__ gamma,
    const float* __restrict__ beta, unsigned short* __restrict__ out) {
    __shared__ float red[256];
    int row = blockIdx.x;
    int tid = threadIdx.x;
    const float* xr = x + (size_t)row * Dc;

    float s = 0.f;
    for (int i = tid; i < Dc; i += 256) s += xr[i];
    red[tid] = s; __syncthreads();
    for (int st = 128; st > 0; st >>= 1) {
        if (tid < st) red[tid] += red[tid + st];
        __syncthreads();
    }
    float mean = red[0] / (float)Dc;
    __syncthreads();

    float v = 0.f;
    for (int i = tid; i < Dc; i += 256) { float d = xr[i] - mean; v += d * d; }
    red[tid] = v; __syncthreads();
    for (int st = 128; st > 0; st >>= 1) {
        if (tid < st) red[tid] += red[tid + st];
        __syncthreads();
    }
    float rstd = rsqrtf(red[0] / (float)Dc + 1e-6f);

    for (int i = tid; i < Dc; i += 256) {
        float y = gamma[i] * (xr[i] - mean) * rstd + beta[i];
        out[(size_t)row * Dc + i] = f32_to_bf16(y);
    }
}

// ---------------------------------------------------------------------------
// Generic WMMA GEMM: C[M,N] = A[M,K] (bf16, row-major) x B[K,N] given as
// BT[N,K] (bf16).  Block = 8 waves (2 row-groups x 4 col-groups); each wave
// owns a 32x64 strip: 2 A fragments + 4 B fragments -> 8 wmma per K-step.
// Epilogues select output transform.
// ---------------------------------------------------------------------------
enum { EPI_BF16 = 0, EPI_VT = 1, EPI_RES = 2, EPI_GELU = 3, EPI_BRES = 4 };

template <int EPI, int M, int N, int K>
__global__ __launch_bounds__(256) void gemm_bf16_wmma(
    const unsigned short* __restrict__ A, const unsigned short* __restrict__ BT,
    void* __restrict__ OutP, const float* __restrict__ Res,
    const float* __restrict__ Bias) {
    int w    = threadIdx.x >> 5;
    int lane = threadIdx.x & 31;
    int row0 = blockIdx.y * 64 + (w & 1) * 32;    // 2 row-groups of 32
    int col0 = blockIdx.x * 256 + (w >> 1) * 64;  // 4 col-groups of 64

    v8f zero = {0.f, 0.f, 0.f, 0.f, 0.f, 0.f, 0.f, 0.f};
    v8f acc0[4] = {zero, zero, zero, zero};
    v8f acc1[4] = {zero, zero, zero, zero};

    for (int k0 = 0; k0 < K; k0 += 32) {
        // prefetch next K-step cachelines (lowers to global_prefetch_b8)
        if (k0 + 32 < K) {
            __builtin_prefetch(A + (size_t)(row0 + (lane & 15)) * K + k0 + 32, 0, 1);
            __builtin_prefetch(BT + (size_t)(col0 + (lane & 15)) * K + k0 + 32, 0, 1);
        }
        v16bf a0 = load_frag(A, K, row0, k0);
        v16bf a1 = load_frag(A, K, row0 + 16, k0);
#pragma unroll
        for (int t = 0; t < 4; ++t) {
            v16bf bfr = load_frag(BT, K, col0 + t * 16, k0);
            acc0[t] = wmma_bf16(a0, bfr, acc0[t]);
            acc1[t] = wmma_bf16(a1, bfr, acc1[t]);
        }
    }

    int cL = lane & 15, hiL = lane >> 4;
#pragma unroll
    for (int rt = 0; rt < 2; ++rt) {
#pragma unroll
        for (int t = 0; t < 4; ++t) {
#pragma unroll
            for (int e = 0; e < 8; ++e) {
                int row = row0 + rt * 16 + e + hiL * 8;
                int col = col0 + t * 16 + cL;
                float v = (rt == 0) ? acc0[t][e] : acc1[t][e];
                if constexpr (EPI == EPI_BF16) {
                    ((unsigned short*)OutP)[(size_t)row * N + col] = f32_to_bf16(v);
                } else if constexpr (EPI == EPI_VT) {
                    // out is V^T [B, H, DK, S]; row = b*S + s, col = h*DK + dk
                    int bb = row >> 11, s = row & (Sc - 1);
                    int hh = col >> 6,  dk = col & (DKc - 1);
                    ((unsigned short*)OutP)[(((size_t)(bb * Hc + hh)) * DKc + dk) * Sc + s] =
                        f32_to_bf16(v);
                } else if constexpr (EPI == EPI_RES) {
                    ((float*)OutP)[(size_t)row * N + col] = v + Res[(size_t)row * N + col];
                } else if constexpr (EPI == EPI_GELU) {
                    float xx = v + Bias[col];
                    float gl = 0.5f * xx * (1.0f + erff(xx * 0.70710678118654752f));
                    ((unsigned short*)OutP)[(size_t)row * N + col] = f32_to_bf16(gl);
                } else { // EPI_BRES
                    ((float*)OutP)[(size_t)row * N + col] =
                        v + Bias[col] + Res[(size_t)row * N + col];
                }
            }
        }
    }
}

// ---------------------------------------------------------------------------
// Attention: one wave per (b, h, 16-query tile).  Two-pass softmax with score
// recompute; QK^T and P*V both via v_wmma_f32_16x16x32_bf16.
// ---------------------------------------------------------------------------
__global__ __launch_bounds__(32) void attn_kernel(
    const unsigned short* __restrict__ q, const unsigned short* __restrict__ k,
    const unsigned short* __restrict__ vT, const int* __restrict__ mask,
    unsigned short* __restrict__ attn) {
    __shared__ __align__(16) float          tileS[16 * 16];
    __shared__ __align__(16) unsigned short P[16 * 32];
    __shared__ float m_s[16];
    __shared__ float l_s[16];

    int bid = blockIdx.x;
    int qt  = bid & 127;          // S/16 = 128
    int h   = (bid >> 7) & 15;
    int b   = bid >> 11;
    int lane = threadIdx.x;
    int cL = lane & 15, hiL = lane >> 4;

    const unsigned short* qb = q  + (size_t)b * Sc * Dc + h * DKc;
    const unsigned short* kb = k  + (size_t)b * Sc * Dc + h * DKc;
    const unsigned short* vb = vT + ((size_t)(b * Hc + h)) * DKc * Sc;
    const int* mb = mask + (size_t)b * Sc;
    int qbase = qt * 16;

    v16bf aq0 = load_frag(qb, Dc, qbase, 0);
    v16bf aq1 = load_frag(qb, Dc, qbase, 32);
    v8f zero = {0.f, 0.f, 0.f, 0.f, 0.f, 0.f, 0.f, 0.f};

    // ---- Pass 1: row max of scaled+masked scores ----
    float m = -1.0e30f;
    for (int kt = 0; kt < 128; ++kt) {
        int key0 = kt * 16;
        v16bf bk0 = load_frag(kb, Dc, key0, 0);
        v16bf bk1 = load_frag(kb, Dc, key0, 32);
        v8f c = zero;
        c = wmma_bf16(aq0, bk0, c);
        c = wmma_bf16(aq1, bk1, c);
        bool mz = (mb[key0 + cL] == 0);
#pragma unroll
        for (int e = 0; e < 8; ++e) {
            float sc = c[e] * 0.125f;           // 1/sqrt(64)
            if (mz) sc = -1.0e9f;
            tileS[(e + hiL * 8) * 16 + cL] = sc;
        }
        __syncthreads();
        if (lane < 16) {
#pragma unroll
            for (int j = 0; j < 16; ++j) m = fmaxf(m, tileS[lane * 16 + j]);
        }
        __syncthreads();
    }
    if (lane < 16) m_s[lane] = m;
    __syncthreads();

    // ---- Pass 2: P = exp(s - m) (unnormalized), O = P*V, l = row sums ----
    float l = 0.f;
    v8f o[4] = {zero, zero, zero, zero};
    for (int kt = 0; kt < 64; ++kt) {
        int key0 = kt * 32;
#pragma unroll
        for (int hf = 0; hf < 2; ++hf) {
            int kbase = key0 + hf * 16;
            v16bf bk0 = load_frag(kb, Dc, kbase, 0);
            v16bf bk1 = load_frag(kb, Dc, kbase, 32);
            v8f c = zero;
            c = wmma_bf16(aq0, bk0, c);
            c = wmma_bf16(aq1, bk1, c);
            bool mz = (mb[kbase + cL] == 0);
#pragma unroll
            for (int e = 0; e < 8; ++e) {
                int r = e + hiL * 8;
                float sc = c[e] * 0.125f;
                if (mz) sc = -1.0e9f;
                float p = __expf(sc - m_s[r]);
                P[r * 32 + hf * 16 + cL] = f32_to_bf16(p);
            }
        }
        __syncthreads();
        if (lane < 16) {
            float s = 0.f;
#pragma unroll
            for (int j = 0; j < 32; ++j) s += bf16_to_f32(P[lane * 32 + j]);
            l += s;
        }
        __syncthreads();
        v16bf ap = load_frag(P, 32, 0, 0);
#pragma unroll
        for (int t = 0; t < 4; ++t) {
            v16bf bv = load_frag(vb, Sc, t * 16, key0);
            o[t] = wmma_bf16(ap, bv, o[t]);
        }
        __syncthreads();
    }
    if (lane < 16) l_s[lane] = l;
    __syncthreads();

    unsigned short* ob = attn + (size_t)b * Sc * Dc + h * DKc;
#pragma unroll
    for (int t = 0; t < 4; ++t) {
#pragma unroll
        for (int e = 0; e < 8; ++e) {
            int r = e + hiL * 8;
            float val = o[t][e] / l_s[r];
            ob[(size_t)(qbase + r) * Dc + t * 16 + cL] = f32_to_bf16(val);
        }
    }
}

// ---------------------------------------------------------------------------
// Host-side launch
// ---------------------------------------------------------------------------
extern "C" void kernel_launch(void* const* d_in, const int* in_sizes, int n_in,
                              void* d_out, int out_size, void* d_ws, size_t ws_size,
                              hipStream_t stream) {
    (void)in_sizes; (void)n_in; (void)out_size; (void)ws_size;
    const float* X    = (const float*)d_in[0];
    const int*   mask = (const int*)d_in[1];
    const float* W_Q  = (const float*)d_in[2];
    const float* W_K  = (const float*)d_in[3];
    const float* W_V  = (const float*)d_in[4];
    const float* W_O  = (const float*)d_in[5];
    const float* g1   = (const float*)d_in[6];
    const float* b1   = (const float*)d_in[7];
    const float* W1   = (const float*)d_in[8];
    const float* bias1= (const float*)d_in[9];
    const float* W2   = (const float*)d_in[10];
    const float* bias2= (const float*)d_in[11];
    const float* g2   = (const float*)d_in[12];
    const float* b2   = (const float*)d_in[13];

    char* ws = (char*)d_ws;
    // bf16 buffers (bytes)
    const size_t szDD   = (size_t)Dc * Dc * 2;          // 2 MB
    const size_t szD4D  = (size_t)Dc * D4 * 2;          // 8 MB
    const size_t szAct  = (size_t)Mrows * Dc * 2;       // 8 MB
    const size_t szActF = (size_t)Mrows * Dc * 4;       // 16 MB

    unsigned short* WqT = (unsigned short*)(ws);
    unsigned short* WkT = (unsigned short*)(ws + szDD);
    unsigned short* WvT = (unsigned short*)(ws + 2 * szDD);
    unsigned short* WoT = (unsigned short*)(ws + 3 * szDD);
    unsigned short* W1T = (unsigned short*)(ws + 4 * szDD);
    unsigned short* W2T = (unsigned short*)(ws + 4 * szDD + szD4D);
    char* p = ws + 4 * szDD + 2 * szD4D;
    unsigned short* xn   = (unsigned short*)(p);              p += szAct;
    unsigned short* qb   = (unsigned short*)(p);              p += szAct;
    unsigned short* kbuf = (unsigned short*)(p);              p += szAct;
    unsigned short* vTb  = (unsigned short*)(p);              p += szAct;
    unsigned short* attn = (unsigned short*)(p);              p += szAct;
    float*          res1 = (float*)(p);                       p += szActF;
    unsigned short* nn2  = (unsigned short*)(p);              p += szAct;
    unsigned short* hbuf = (unsigned short*)(p);              /* 32 MB */

    // 1) Weight conversion + transpose to bf16
    {
        int nDD = Dc * Dc;
        convT_kernel<<<(nDD + 255) / 256, 256, 0, stream>>>(W_Q, WqT, Dc, Dc);
        convT_kernel<<<(nDD + 255) / 256, 256, 0, stream>>>(W_K, WkT, Dc, Dc);
        convT_kernel<<<(nDD + 255) / 256, 256, 0, stream>>>(W_V, WvT, Dc, Dc);
        convT_kernel<<<(nDD + 255) / 256, 256, 0, stream>>>(W_O, WoT, Dc, Dc);
        int nD4 = Dc * D4;
        convT_kernel<<<(nD4 + 255) / 256, 256, 0, stream>>>(W1, W1T, Dc, D4);
        convT_kernel<<<(nD4 + 255) / 256, 256, 0, stream>>>(W2, W2T, D4, Dc);
    }

    // 2) LayerNorm 1
    layernorm_bf16_kernel<<<Mrows, 256, 0, stream>>>(X, g1, b1, xn);

    // 3) Q, K projections; V projection with transposed [B,H,DK,S] output
    gemm_bf16_wmma<EPI_BF16, Mrows, Dc, Dc>
        <<<dim3(Dc / 256, Mrows / 64), 256, 0, stream>>>(xn, WqT, qb, nullptr, nullptr);
    gemm_bf16_wmma<EPI_BF16, Mrows, Dc, Dc>
        <<<dim3(Dc / 256, Mrows / 64), 256, 0, stream>>>(xn, WkT, kbuf, nullptr, nullptr);
    gemm_bf16_wmma<EPI_VT, Mrows, Dc, Dc>
        <<<dim3(Dc / 256, Mrows / 64), 256, 0, stream>>>(xn, WvT, vTb, nullptr, nullptr);

    // 4) Attention (B*H*(S/16) = 4096 waves)
    attn_kernel<<<Bc * Hc * (Sc / 16), 32, 0, stream>>>(qb, kbuf, vTb, mask, attn);

    // 5) Output projection + residual (f32)
    gemm_bf16_wmma<EPI_RES, Mrows, Dc, Dc>
        <<<dim3(Dc / 256, Mrows / 64), 256, 0, stream>>>(attn, WoT, res1, X, nullptr);

    // 6) LayerNorm 2
    layernorm_bf16_kernel<<<Mrows, 256, 0, stream>>>(res1, g2, b2, nn2);

    // 7) FFN1 + bias + GELU (bf16 out)
    gemm_bf16_wmma<EPI_GELU, Mrows, D4, Dc>
        <<<dim3(D4 / 256, Mrows / 64), 256, 0, stream>>>(nn2, W1T, hbuf, nullptr, bias1);

    // 8) FFN2 + bias + residual -> d_out (f32)
    gemm_bf16_wmma<EPI_BRES, Mrows, Dc, D4>
        <<<dim3(Dc / 256, Mrows / 64), 256, 0, stream>>>(hbuf, W2T, (float*)d_out,
                                                         res1, bias2);
}
